// TestModule_15453292331515
// MI455X (gfx1250) — matrix operand order repaired and checked by reading, not compiled, not run on hardware
//
#include <hip/hip_runtime.h>
#include <hip/hip_bf16.h>
#include <stdint.h>

// ---- shapes (fixed by the reference) ----
#define G_  16
#define S_  2048
#define H_  1024
#define F_  4096

// ---- tiling ----
#define BM  128          // block tile rows (M)
#define BN  128          // block tile cols (N)
#define BK  32           // k-step (bf16 WMMA K)
#define LDA 40           // BK + 8 pad (bf16 elems) -> 80B row stride, conflict-free b128 reads
#define NKT (H_ / BK)    // 32 k-steps

typedef __attribute__((ext_vector_type(16))) __bf16 v16bf;
typedef __attribute__((ext_vector_type(8)))  __bf16 v8bf;
typedef __attribute__((ext_vector_type(4)))  __bf16 v4bf;
typedef __attribute__((ext_vector_type(8)))  float  v8f;

static __device__ __forceinline__ v16bf cat16(v8bf a, v8bf b) {
  return __builtin_shufflevector(a, b, 0,1,2,3,4,5,6,7,8,9,10,11,12,13,14,15);
}

__global__ __launch_bounds__(256)
void grouped_gemm_bf16x3(const float* __restrict__ hs,
                         const float* __restrict__ wt,
                         const int*   __restrict__ seq_lens,
                         float*       __restrict__ out) {
  // hi/lo bf16 split tiles, double buffered. A: [M][K], B stored transposed: [N][K].
  // 2*(128+128)*40*2B*2(hi,lo) = 80 KB of LDS.
  __shared__ __bf16 Ah[2][BM][LDA];
  __shared__ __bf16 Al[2][BM][LDA];
  __shared__ __bf16 Bh[2][BN][LDA];
  __shared__ __bf16 Bl[2][BN][LDA];

  const int tid = threadIdx.x;
  const int g   = blockIdx.z;
  const int s0  = blockIdx.y * BM;
  const int f0  = blockIdx.x * BN;
  const int seqLen = seq_lens[g];

  const int lane   = tid & 31;
  const int w      = tid >> 5;
  const int wm     = w >> 2;      // 0..1 -> 64 rows each
  const int wn     = w & 3;       // 0..3 -> 32 cols each (two 16-wide subtiles)
  const int lane15 = lane & 15;
  const int sel    = lane >> 4;   // half-wave select per ISA fragment layout

  const float* gA = hs + (size_t)g * S_ * H_;
  const float* gB = wt + (size_t)g * H_ * F_;

  float4 aR[4];   // staged A: 128x32 fp32 = 1024 float4 / 256 thr = 4 each
  float4 bR[4];   // staged B: 32x128 fp32 = 1024 float4 / 256 thr = 4 each

  auto loadGlobal = [&](int kt) {
    const int k0 = kt * BK;
#pragma unroll
    for (int i = 0; i < 4; ++i) {
      const int idx = tid + i * 256;        // 0..1023
      const int r   = idx >> 3;             // 0..127
      const int c4  = idx & 7;              // 0..7
      const int s   = s0 + r;
      float4 v = *(const float4*)(gA + (size_t)s * H_ + k0 + c4 * 4);
      if (s >= seqLen) v = make_float4(0.f, 0.f, 0.f, 0.f);  // ragged mask
      aR[i] = v;
    }
#pragma unroll
    for (int i = 0; i < 4; ++i) {
      const int idx = tid + i * 256;        // 0..1023
      const int kk  = idx >> 5;             // 0..31
      const int c4  = idx & 31;             // 0..31
      bR[i] = *(const float4*)(gB + (size_t)(k0 + kk) * F_ + f0 + c4 * 4);
    }
  };

  auto storeLds = [&](int buf) {
#pragma unroll
    for (int i = 0; i < 4; ++i) {
      const int idx = tid + i * 256;
      const int r   = idx >> 3;
      const int c4  = idx & 7;
      const float x[4] = {aR[i].x, aR[i].y, aR[i].z, aR[i].w};
      v4bf hi, lo;
#pragma unroll
      for (int j = 0; j < 4; ++j) {
        const __bf16 h = (__bf16)x[j];
        hi[j] = h;
        lo[j] = (__bf16)(x[j] - (float)h);   // bf16x3 residual
      }
      *(v4bf*)&Ah[buf][r][c4 * 4] = hi;
      *(v4bf*)&Al[buf][r][c4 * 4] = lo;
    }
#pragma unroll
    for (int i = 0; i < 4; ++i) {
      const int idx = tid + i * 256;
      const int kk  = idx >> 5;
      const int c4  = idx & 31;
      const float x[4] = {bR[i].x, bR[i].y, bR[i].z, bR[i].w};
#pragma unroll
      for (int j = 0; j < 4; ++j) {          // transpose into [N][K]
        const __bf16 h = (__bf16)x[j];
        Bh[buf][c4 * 4 + j][kk] = h;
        Bl[buf][c4 * 4 + j][kk] = (__bf16)(x[j] - (float)h);
      }
    }
  };

  v8f acc[4][2] = {};   // 4 (M) x 2 (N) 16x16 f32 C tiles per wave

  loadGlobal(0);
  storeLds(0);
  __syncthreads();

  int buf = 0;
#pragma unroll 1
  for (int kt = 0; kt < NKT; ++kt) {
    if (kt + 1 < NKT) loadGlobal(kt + 1);
    if (kt + 2 < NKT) {  // pull k+2 toward L2 (global_prefetch_b8)
      const int k2 = (kt + 2) * BK;
      __builtin_prefetch(gA + (size_t)(s0 + (tid >> 3)) * H_ + k2, 0, 1);
      __builtin_prefetch(gB + (size_t)(k2 + (tid >> 5)) * F_ + f0, 0, 1);
    }

    // B fragments (this wave's two 16-wide N slices), contiguous K order (same as A):
    // lanes 0-15: K=0..15 of column N=lane; lanes 16-31: K=16..31.
    v16bf bh[2], bl[2];
#pragma unroll
    for (int bn = 0; bn < 2; ++bn) {
      const __bf16* bhRow = &Bh[buf][wn * 32 + bn * 16 + lane15][0];
      const __bf16* blRow = &Bl[buf][wn * 32 + bn * 16 + lane15][0];
      bh[bn] = cat16(*(const v8bf*)(bhRow + sel * 16),
                     *(const v8bf*)(bhRow + sel * 16 + 8));
      bl[bn] = cat16(*(const v8bf*)(blRow + sel * 16),
                     *(const v8bf*)(blRow + sel * 16 + 8));
    }

#pragma unroll
    for (int am = 0; am < 4; ++am) {
      // A fragment, same K order as B: lanes 0-15 hold K 0..15,
      // lanes 16-31 hold K 16..31 (two adjacent 16B chunks).
      const __bf16* ahRow = &Ah[buf][wm * 64 + am * 16 + lane15][0];
      const __bf16* alRow = &Al[buf][wm * 64 + am * 16 + lane15][0];
      const v16bf ah = cat16(*(const v8bf*)(ahRow + sel * 16),
                             *(const v8bf*)(ahRow + sel * 16 + 8));
      const v16bf al = cat16(*(const v8bf*)(alRow + sel * 16),
                             *(const v8bf*)(alRow + sel * 16 + 8));

#pragma unroll
      for (int bn = 0; bn < 2; ++bn) {
        // bf16x3: D += aHi*bHi + aHi*bLo + aLo*bHi  (aLo*bLo dropped, ~2^-16 level)
        acc[am][bn] = __builtin_amdgcn_wmma_f32_16x16x32_bf16(
            false, ah, false, bh[bn], (short)0, acc[am][bn], false, false);
        acc[am][bn] = __builtin_amdgcn_wmma_f32_16x16x32_bf16(
            false, ah, false, bl[bn], (short)0, acc[am][bn], false, false);
        acc[am][bn] = __builtin_amdgcn_wmma_f32_16x16x32_bf16(
            false, al, false, bh[bn], (short)0, acc[am][bn], false, false);
      }
    }

    if (kt + 1 < NKT) storeLds(buf ^ 1);
    __syncthreads();
    buf ^= 1;
  }

  // C layout: VGPR j -> (lanes 0-15: M=j, N=lane), (lanes 16-31: M=8+j, N=lane-16).
  float* gOut = out + (size_t)g * S_ * F_;
#pragma unroll
  for (int am = 0; am < 4; ++am) {
    const int srow = s0 + wm * 64 + am * 16 + sel * 8;
#pragma unroll
    for (int bn = 0; bn < 2; ++bn) {
      const int fcol = f0 + wn * 32 + bn * 16 + lane15;
#pragma unroll
      for (int j = 0; j < 8; ++j) {
        gOut[(size_t)(srow + j) * F_ + fcol] = acc[am][bn][j];
      }
    }
  }
}

extern "C" void kernel_launch(void* const* d_in, const int* in_sizes, int n_in,
                              void* d_out, int out_size, void* d_ws, size_t ws_size,
                              hipStream_t stream) {
  (void)in_sizes; (void)n_in; (void)out_size; (void)d_ws; (void)ws_size;
  const float* hs  = (const float*)d_in[0];   // [G,S,H] fp32
  const float* wtp = (const float*)d_in[1];   // [G,H,F] fp32
  const int*   sl  = (const int*)d_in[2];     // [G] int32
  float*       out = (float*)d_out;           // [G,S,F] fp32

  dim3 grid(F_ / BN, S_ / BM, G_);            // 32 x 16 x 16 workgroups
  grouped_gemm_bf16x3<<<grid, 256, 0, stream>>>(hs, wtp, sl, out);
}
